// GEncoder_12163347383179
// MI455X (gfx1250) — matrix operand order repaired and checked by reading, not compiled, not run on hardware
//
#include <hip/hip_runtime.h>
#include <hip/hip_bf16.h>

// ---------------------------------------------------------------------------
// GNN encoder for MI455X (gfx1250). All dense GEMMs run on WMMA
// v_wmma_f32_16x16x32_f16 (f16 in / f32 accumulate) with double-buffered LDS
// staging. Weights are pre-transposed at f32->f16 conversion so both A and B
// tiles stage with b128 loads/stores (no scalar transpose stores). q/k
// projections are algebraically folded into rank-4 per-head vectors.
// Segment softmax + aggregation via f32 atomics.
// ---------------------------------------------------------------------------

typedef _Float16 half_t;
typedef __attribute__((ext_vector_type(16))) _Float16 v16h;
typedef __attribute__((ext_vector_type(8)))  float    v8f;

#define BM 128
#define BN 128
#define BK 32
#define LDSPAD 8
#define EPSRMS 1.1920929e-07f

// -------- WMMA GEMM: C[M,N] = A16[M,K] @ (Bt16[N,K])^T + bias[N] -----------
__global__ __launch_bounds__(256)
void gemm_f16_wmma(const half_t* __restrict__ A, const half_t* __restrict__ Bt,
                   const float* __restrict__ bias, float* __restrict__ C,
                   int M, int N, int K)
{
  __shared__ __align__(16) half_t As[2][BM][BK + LDSPAD];
  __shared__ __align__(16) half_t Bs[2][BN][BK + LDSPAD];   // Bs[n][k]

  const int tid  = threadIdx.x;
  const int lane = tid & 31;
  const int wid  = tid >> 5;
  const int wm   = wid >> 2;   // 0..1 -> 64-row slab
  const int wn   = wid & 3;    // 0..3 -> 32-col slab
  const int rowBase = blockIdx.y * BM;
  const int colBase = blockIdx.x * BN;

  // staging map: 128 tile rows x 32 k's; 2 threads/row, 16 halves each
  const int lr = tid >> 1;             // 0..127
  const int lc = (tid & 1) * 16;       // 0 or 16

  const int  ga_row = rowBase + lr;
  const bool a_ok   = ga_row < M;
  const half_t* gA = A  + (size_t)ga_row * K + lc;            // + kb
  const half_t* gB = Bt + (size_t)(colBase + lr) * K + lc;    // + kb

  const int nk = K / BK;
  v8f acc[4][2] = {};

  uint4 ra0 = make_uint4(0u,0u,0u,0u), ra1 = ra0, rb0, rb1;

  // ---- prologue: stage tile 0 ----
  if (a_ok) { const uint4* p = (const uint4*)gA; ra0 = p[0]; ra1 = p[1]; }
  { const uint4* p = (const uint4*)gB; rb0 = p[0]; rb1 = p[1]; }
  { uint4* s = (uint4*)&As[0][lr][lc]; s[0] = ra0; s[1] = ra1; }
  { uint4* s = (uint4*)&Bs[0][lr][lc]; s[0] = rb0; s[1] = rb1; }
  __syncthreads();

  for (int t = 0; t < nk; ++t) {
    const int cur = t & 1, nxt = cur ^ 1;

    // ---- issue next tile's global loads (overlaps with WMMA below) ----
    if (t + 1 < nk) {
      const int kb = (t + 1) * BK;
      ra0 = make_uint4(0u,0u,0u,0u); ra1 = ra0;
      if (a_ok) { const uint4* p = (const uint4*)(gA + kb); ra0 = p[0]; ra1 = p[1]; }
      { const uint4* p = (const uint4*)(gB + kb); rb0 = p[0]; rb1 = p[1]; }
      if (t + 2 < nk) {
        __builtin_prefetch(gA + (size_t)(t + 2) * BK, 0, 0);
        __builtin_prefetch(gB + (size_t)(t + 2) * BK, 0, 0);
      }
    }

    // ---- fragments per ISA 7.12.2 layouts ----
    v16h af[4];
    {
      const int r0 = wm * 64 + (lane & 15);
      const int ka = (lane < 16) ? 0 : 8;      // K pairs {0-7,16-23} vs {8-15,24-31}
#pragma unroll
      for (int tm = 0; tm < 4; ++tm) {
        const half_t* p = &As[cur][r0 + tm * 16][ka];
        uint4* v = (uint4*)&af[tm];
        v[0] = *(const uint4*)p;
        v[1] = *(const uint4*)(p + 16);
      }
    }
    v16h bf[2];
    {
      const int c0 = wn * 32 + (lane & 15);
      const int kB = (lane < 16) ? 0 : 8;      // K pairs {0-7,16-23} vs {8-15,24-31}
#pragma unroll
      for (int tn = 0; tn < 2; ++tn) {
        const half_t* p = &Bs[cur][c0 + tn * 16][kB];
        uint4* v = (uint4*)&bf[tn];
        v[0] = *(const uint4*)p;
        v[1] = *(const uint4*)(p + 16);
      }
    }
#pragma unroll
    for (int tm = 0; tm < 4; ++tm)
#pragma unroll
      for (int tn = 0; tn < 2; ++tn)
        acc[tm][tn] = __builtin_amdgcn_wmma_f32_16x16x32_f16(
            false, af[tm], false, bf[tn], (short)0, acc[tm][tn], false, false);

    // ---- commit next tile into the other buffer ----
    if (t + 1 < nk) {
      uint4* sa = (uint4*)&As[nxt][lr][lc]; sa[0] = ra0; sa[1] = ra1;
      uint4* sb = (uint4*)&Bs[nxt][lr][lc]; sb[0] = rb0; sb[1] = rb1;
    }
    __syncthreads();
  }

  // ---- epilogue: C layout VGPR i -> M=i (+8 for upper half-wave) ----
  const int rlo = (lane < 16) ? 0 : 8;
  const int cl  = lane & 15;
#pragma unroll
  for (int tm = 0; tm < 4; ++tm) {
#pragma unroll
    for (int tn = 0; tn < 2; ++tn) {
      const int col = colBase + wn * 32 + tn * 16 + cl;
      const float bv = bias ? bias[col] : 0.f;
#pragma unroll
      for (int i = 0; i < 8; ++i) {
        const int row = rowBase + wm * 64 + tm * 16 + rlo + i;
        if (row < M) C[(size_t)row * N + col] = acc[tm][tn][i] + bv;
      }
    }
  }
}

// ---------------- helpers ---------------------------------------------------
__device__ __forceinline__ void atomicMaxF(float* a, float v) {
  if (v >= 0.f) atomicMax((int*)a, __float_as_int(v));
  else          atomicMin((unsigned int*)a, __float_as_uint(v));
}

__device__ __forceinline__ float blockReduceSum(float v, float* sh) {
  const int tid = threadIdx.x;
  sh[tid] = v; __syncthreads();
  for (int s = blockDim.x >> 1; s > 0; s >>= 1) {
    if (tid < s) sh[tid] += sh[tid + s];
    __syncthreads();
  }
  float r = sh[0]; __syncthreads();
  return r;
}

// ---------------- embedding encoders ---------------------------------------
__global__ void encode_nodes(const int* __restrict__ x, const float* __restrict__ emb,
                             float* __restrict__ h, half_t* __restrict__ h16) {
  const int i = blockIdx.x * 256 + threadIdx.x;   // N*512
  const int n = i >> 9, c = i & 511;
  const int offs[9] = {0,119,128,139,151,160,165,173,175};
  float s = 0.f;
#pragma unroll
  for (int j = 0; j < 9; ++j)
    s += emb[(size_t)(x[n * 9 + j] + offs[j]) * 512 + c];
  h[i] = s; h16[i] = (half_t)s;
}

__global__ void encode_edges(const int* __restrict__ ea, const float* __restrict__ emb,
                             half_t* __restrict__ e16) {
  const int i = blockIdx.x * 256 + threadIdx.x;   // E*128
  const int e = i >> 7, c = i & 127;
  const int offs[3] = {0,119,128};
  float s = 0.f;
#pragma unroll
  for (int j = 0; j < 3; ++j)
    s += emb[(size_t)(ea[e * 3 + j] + offs[j]) * 128 + c];
  e16[i] = (half_t)s;
}

// ---------------- folded attention: Weff[d,hh] = Wq[d,hh*128:]·Watt --------
__global__ void fold_qk(const float* __restrict__ Wq, const float* __restrict__ Wk,
                        const float* __restrict__ bq, const float* __restrict__ bk,
                        const float* __restrict__ Watt,
                        float* __restrict__ weffq, float* __restrict__ weffk,
                        float* __restrict__ beff, int l) {
  const int t = blockIdx.x * 256 + threadIdx.x;
  if (t < 4096) {
    const int d = t >> 3, sel = t & 7, which = sel >> 2, hh = sel & 3;
    const float* W  = (which ? Wk : Wq) + (size_t)l * 512 * 512 + (size_t)d * 512 + hh * 128;
    const float* wa = Watt + l * 256 + which * 128;
    float s = 0.f;
    for (int j = 0; j < 128; ++j) s += W[j] * wa[j];
    (which ? weffk : weffq)[d * 4 + hh] = s;
  } else if (t < 4104) {
    const int sel = t - 4096, which = sel >> 2, hh = sel & 3;
    const float* bb = (which ? bk : bq) + l * 512 + hh * 128;
    const float* wa = Watt + l * 256 + which * 128;
    float s = 0.f;
    for (int j = 0; j < 128; ++j) s += bb[j] * wa[j];
    beff[sel] = s;
  }
}

__global__ void qa_ka(const float* __restrict__ h,
                      const float* __restrict__ weffq, const float* __restrict__ weffk,
                      const float* __restrict__ beff,
                      float* __restrict__ qa, float* __restrict__ ka) {
  const int t = blockIdx.x * 256 + threadIdx.x;   // N*8
  const int n = t >> 3, sel = t & 7, which = sel >> 2, hh = sel & 3;
  const float* W  = which ? weffk : weffq;
  const float* hr = h + (size_t)n * 512;
  float s = beff[sel];
  for (int d = 0; d < 512; ++d) s += hr[d] * W[d * 4 + hh];
  (which ? ka : qa)[n * 4 + hh] = s;
}

// ---------------- per-layer init + segment softmax --------------------------
__global__ void layer_init(float* smax, float* ssum, float* aggr, int N) {
  const int i = blockIdx.x * 256 + threadIdx.x;   // N*512
  if (i < N * 4) { ((unsigned int*)smax)[i] = 0xFF800000u; ssum[i] = 0.f; }
  aggr[i] = 0.f;
}

__global__ void score_max(const float* __restrict__ qa, const float* __restrict__ ka,
                          const int* __restrict__ src, const int* __restrict__ dst,
                          float* __restrict__ sbuf, float* __restrict__ smax) {
  const int i = blockIdx.x * 256 + threadIdx.x;   // E*4
  const int e = i >> 2, hh = i & 3;
  const int d = dst[e], s = src[e];
  float v = qa[d * 4 + hh] + ka[s * 4 + hh];
  v = (v >= 0.f) ? v : 0.05f * v;                 // LeakyReLU(0.05)
  sbuf[i] = v;
  atomicMaxF(&smax[d * 4 + hh], v);
}

__global__ void exp_sum(float* __restrict__ sbuf, const float* __restrict__ smax,
                        float* __restrict__ ssum, const int* __restrict__ dst) {
  const int i = blockIdx.x * 256 + threadIdx.x;   // E*4
  const int e = i >> 2, hh = i & 3;
  const float ex = expf(sbuf[i] - smax[dst[e] * 4 + hh]);
  sbuf[i] = ex;
  atomicAdd(&ssum[dst[e] * 4 + hh], ex);
}

// ---------------- message assembly / scatter --------------------------------
__global__ void gather_mi(const half_t* __restrict__ h16, const half_t* __restrict__ e16,
                          const int* __restrict__ src, half_t* __restrict__ A) {
  const long i = (long)blockIdx.x * 256 + threadIdx.x;  // E*640
  const int e = (int)(i / 640), c = (int)(i % 640);
  A[i] = (c < 512) ? h16[(size_t)src[e] * 512 + c]
                   : e16[(size_t)e * 128 + (c - 512)];
}

__global__ void scatter_msg(const float* __restrict__ m, const float* __restrict__ sbuf,
                            const float* __restrict__ ssum, const int* __restrict__ dst,
                            float* __restrict__ aggr) {
  const long i = (long)blockIdx.x * 256 + threadIdx.x;  // E*512
  const int e = (int)(i >> 9), c = (int)(i & 511), hh = c >> 7;
  const int d = dst[e];
  const float alpha = sbuf[e * 4 + hh] / (ssum[d * 4 + hh] + 1e-16f);
  atomicAdd(&aggr[(size_t)d * 512 + c], m[i] * alpha);
}

__global__ void build_upd(const half_t* __restrict__ h16, const float* __restrict__ aggr,
                          half_t* __restrict__ A) {
  const int i = blockIdx.x * 256 + threadIdx.x;   // N*1024
  const int n = i >> 10, c = i & 1023;
  A[i] = (c < 512) ? h16[(size_t)n * 512 + c]
                   : (half_t)aggr[(size_t)n * 512 + (c - 512)];
}

// ---------------- norms -----------------------------------------------------
__global__ void rmsnorm_relu_f16(const float* __restrict__ X, const float* __restrict__ g,
                                 half_t* __restrict__ Y, int C) {
  __shared__ float sh[256];
  const int row = blockIdx.x, tid = threadIdx.x;
  const float* xr = X + (size_t)row * C;
  float ss = 0.f;
  for (int c = tid; c < C; c += 256) { const float v = xr[c]; ss += v * v; }
  const float tot = blockReduceSum(ss, sh);
  const float scale = rsqrtf(tot / (float)C + EPSRMS);
  half_t* yr = Y + (size_t)row * C;
  for (int c = tid; c < C; c += 256) {
    float v = xr[c] * scale * g[c];
    yr[c] = (half_t)(v > 0.f ? v : 0.f);
  }
}

__global__ void update_norm(const float* __restrict__ U, const float* __restrict__ gu,
                            const float* __restrict__ gl,
                            float* __restrict__ h, half_t* __restrict__ h16) {
  __shared__ float sh[256];
  const int row = blockIdx.x, tid = threadIdx.x;
  const float* ur = U + (size_t)row * 512;
  const float x0 = ur[tid], x1 = ur[tid + 256];
  const float t1 = blockReduceSum(x0 * x0 + x1 * x1, sh);
  const float s1 = rsqrtf(t1 / 512.f + EPSRMS);
  float y0 = x0 * s1 * gu[tid];       y0 = y0 > 0.f ? y0 : 0.f;
  float y1 = x1 * s1 * gu[tid + 256]; y1 = y1 > 0.f ? y1 : 0.f;
  const float t2 = blockReduceSum(y0 * y0 + y1 * y1, sh);
  const float s2 = rsqrtf(t2 / 512.f + EPSRMS);
  const float z0 = y0 * s2 * gl[tid], z1 = y1 * s2 * gl[tid + 256];
  h[(size_t)row * 512 + tid] = z0;       h[(size_t)row * 512 + tid + 256] = z1;
  h16[(size_t)row * 512 + tid] = (half_t)z0;
  h16[(size_t)row * 512 + tid + 256] = (half_t)z1;
}

// ---------------- pooling head ---------------------------------------------
__global__ void att_softmax(const float* __restrict__ z, const float* __restrict__ Wa,
                            const float* __restrict__ ba, float* __restrict__ asoft) {
  __shared__ float sh[128];
  const int b = blockIdx.x, n = threadIdx.x;   // 128 threads
  const float* zr = z + ((size_t)b * 128 + n) * 512;
  float s = ba[0];
  for (int c = 0; c < 512; ++c) s += zr[c] * Wa[c];
  sh[n] = s; __syncthreads();
  for (int st = 64; st > 0; st >>= 1) { if (n < st) sh[n] = fmaxf(sh[n], sh[n + st]); __syncthreads(); }
  const float mx = sh[0]; __syncthreads();
  const float ex = expf(s - mx);
  sh[n] = ex; __syncthreads();
  for (int st = 64; st > 0; st >>= 1) { if (n < st) sh[n] += sh[n + st]; __syncthreads(); }
  asoft[b * 128 + n] = ex / sh[0];
}

__global__ void pool(const float* __restrict__ z, const float* __restrict__ asoft,
                     half_t* __restrict__ hg16) {
  const int b = blockIdx.x >> 2;
  const int c = ((blockIdx.x & 3) << 7) + threadIdx.x;   // 128 threads
  float acc = 0.f;
  for (int n = 0; n < 128; ++n)
    acc += asoft[b * 128 + n] * z[((size_t)b * 128 + n) * 512 + c];
  hg16[b * 512 + c] = (half_t)acc;
}

// ---- f32 [K,N] weight -> f16 transposed [N,K] ------------------------------
__global__ void convert_w_t(const float* __restrict__ W, half_t* __restrict__ O,
                            int K, int N) {
  const long i = (long)blockIdx.x * 256 + threadIdx.x;  // K*N
  const int n = (int)(i / K), k = (int)(i % K);
  O[i] = (half_t)W[(size_t)k * N + n];
}

// ---------------------------------------------------------------------------
extern "C" void kernel_launch(void* const* d_in, const int* in_sizes, int n_in,
                              void* d_out, int out_size, void* d_ws, size_t ws_size,
                              hipStream_t stream) {
  (void)in_sizes; (void)n_in; (void)out_size; (void)ws_size;
  const int N = 8192, E = 65536, H = 512, HE = 128, L = 3, PD = 1024, B = 64;

  const int*   x         = (const int*)d_in[0];
  const int*   edge_attr = (const int*)d_in[1];
  const int*   edge_index= (const int*)d_in[2];
  const float* emb_node  = (const float*)d_in[3];
  const float* emb_edge  = (const float*)d_in[4];
  const float* Wm1 = (const float*)d_in[5];
  const float* bm1 = (const float*)d_in[6];
  const float* gm  = (const float*)d_in[7];
  const float* Wm2 = (const float*)d_in[8];
  const float* bm2 = (const float*)d_in[9];
  const float* Wq  = (const float*)d_in[10];
  const float* bq  = (const float*)d_in[11];
  const float* Wk  = (const float*)d_in[12];
  const float* bk  = (const float*)d_in[13];
  const float* Watt= (const float*)d_in[14];
  const float* Wu  = (const float*)d_in[15];
  const float* bu  = (const float*)d_in[16];
  const float* gu  = (const float*)d_in[17];
  const float* gln = (const float*)d_in[18];
  const float* Wg  = (const float*)d_in[19];
  const float* bg  = (const float*)d_in[20];
  const float* Wa  = (const float*)d_in[21];
  const float* ba  = (const float*)d_in[22];
  const float* Wp1 = (const float*)d_in[23];
  const float* bp1 = (const float*)d_in[24];
  const float* gp  = (const float*)d_in[25];
  const float* Wp2 = (const float*)d_in[26];
  const float* bp2 = (const float*)d_in[27];
  float* out = (float*)d_out;

  char* ws = (char*)d_ws;
  size_t off = 0;
  auto alloc = [&](size_t bytes) -> char* {
    char* p = ws + off;
    off += (bytes + 255) & ~(size_t)255;
    return p;
  };
  float*  h    = (float*) alloc((size_t)N * H * 4);
  half_t* h16  = (half_t*)alloc((size_t)N * H * 2);
  half_t* e16  = (half_t*)alloc((size_t)E * HE * 2);
  float*  weffq= (float*) alloc(512 * 4 * 4);
  float*  weffk= (float*) alloc(512 * 4 * 4);
  float*  beff = (float*) alloc(8 * 4);
  float*  qa   = (float*) alloc((size_t)N * 4 * 4);
  float*  ka   = (float*) alloc((size_t)N * 4 * 4);
  float*  smax = (float*) alloc((size_t)N * 4 * 4);
  float*  ssum = (float*) alloc((size_t)N * 4 * 4);
  float*  sbuf = (float*) alloc((size_t)E * 4 * 4);
  float*  aggr = (float*) alloc((size_t)N * H * 4);
  half_t* w16  = (half_t*)alloc((size_t)1024 * 1024 * 2);   // transposed weights
  half_t* a16  = (half_t*)alloc((size_t)E * 640 * 2);
  half_t* t16  = (half_t*)alloc((size_t)E * H * 2);
  float*  fbuf = (float*) alloc((size_t)E * H * 4);
  float*  asoft= (float*) alloc((size_t)B * 128 * 4);
  half_t* hg16 = (half_t*)alloc((size_t)B * H * 2);
  float*  pf   = (float*) alloc((size_t)B * PD * 4);
  half_t* p16  = (half_t*)alloc((size_t)B * PD * 2);

  const int T = 256;
  const int* srcI = edge_index;        // row 0
  const int* dstI = edge_index + E;    // row 1

  encode_nodes<<<(N * H) / T, T, 0, stream>>>(x, emb_node, h, h16);
  encode_edges<<<(E * HE) / T, T, 0, stream>>>(edge_attr, emb_edge, e16);

  for (int l = 0; l < L; ++l) {
    // ---- attention scores (folded rank-4 q/k) ----
    fold_qk<<<(4104 + T - 1) / T, T, 0, stream>>>(Wq, Wk, bq, bk, Watt, weffq, weffk, beff, l);
    qa_ka<<<(N * 8) / T, T, 0, stream>>>(h, weffq, weffk, beff, qa, ka);
    layer_init<<<(N * H) / T, T, 0, stream>>>(smax, ssum, aggr, N);
    score_max<<<(E * 4) / T, T, 0, stream>>>(qa, ka, srcI, dstI, sbuf, smax);
    exp_sum<<<(E * 4) / T, T, 0, stream>>>(sbuf, smax, ssum, dstI);

    // ---- message MLP (WMMA GEMMs) ----
    convert_w_t<<<(640 * 512) / T, T, 0, stream>>>(Wm1 + (size_t)l * 640 * 512, w16, 640, 512);
    gather_mi<<<(int)(((size_t)E * 640) / T), T, 0, stream>>>(h16, e16, srcI, a16);
    { dim3 g(512 / BN, (E + BM - 1) / BM);
      gemm_f16_wmma<<<g, T, 0, stream>>>(a16, w16, bm1 + l * 512, fbuf, E, 512, 640); }
    rmsnorm_relu_f16<<<E, T, 0, stream>>>(fbuf, gm + l * 512, t16, 512);
    convert_w_t<<<(512 * 512) / T, T, 0, stream>>>(Wm2 + (size_t)l * 512 * 512, w16, 512, 512);
    { dim3 g(512 / BN, (E + BM - 1) / BM);
      gemm_f16_wmma<<<g, T, 0, stream>>>(t16, w16, bm2 + l * 512, fbuf, E, 512, 512); }
    scatter_msg<<<(int)(((size_t)E * 512) / T), T, 0, stream>>>(fbuf, sbuf, ssum, dstI, aggr);

    // ---- update MLP + double RMSNorm ----
    build_upd<<<(N * 1024) / T, T, 0, stream>>>(h16, aggr, a16);
    convert_w_t<<<(1024 * 512) / T, T, 0, stream>>>(Wu + (size_t)l * 1024 * 512, w16, 1024, 512);
    { dim3 g(512 / BN, N / BM);
      gemm_f16_wmma<<<g, T, 0, stream>>>(a16, w16, bu + l * 512, fbuf, N, 512, 1024); }
    update_norm<<<N, T, 0, stream>>>(fbuf, gu + l * 512, gln + l * 512, h, h16);
  }

  // ---- global attention pooling head ----
  convert_w_t<<<(512 * 512) / T, T, 0, stream>>>(Wg, w16, 512, 512);
  { dim3 g(512 / BN, N / BM);
    gemm_f16_wmma<<<g, T, 0, stream>>>(h16, w16, bg, fbuf, N, 512, 512); }
  att_softmax<<<B, 128, 0, stream>>>(fbuf, Wa, ba, asoft);
  pool<<<B * 4, 128, 0, stream>>>(fbuf, asoft, hg16);

  convert_w_t<<<(512 * 1024) / T, T, 0, stream>>>(Wp1, w16, 512, 1024);
  { dim3 g(1024 / BN, 1);
    gemm_f16_wmma<<<g, T, 0, stream>>>(hg16, w16, bp1, pf, B, 1024, 512); }
  rmsnorm_relu_f16<<<B, T, 0, stream>>>(pf, gp, p16, 1024);
  convert_w_t<<<(1024 * 1024) / T, T, 0, stream>>>(Wp2, w16, 1024, 1024);
  { dim3 g(1024 / BN, 1);
    gemm_f16_wmma<<<g, T, 0, stream>>>(p16, w16, bp2, out, B, 1024, 1024); }
}
